// EncoderBlock_42795054137796
// MI455X (gfx1250) — matrix operand order repaired and checked by reading, not compiled, not run on hardware
//
#include <hip/hip_runtime.h>
#include <hip/hip_bf16.h>

typedef __attribute__((ext_vector_type(16))) __bf16 v16bf;
typedef __attribute__((ext_vector_type(8)))  __bf16 v8bf;
typedef __attribute__((ext_vector_type(8)))  float  v8f;

// CDNA5 async global->LDS path (ASYNCcnt-tracked DMA, bypasses VGPRs).
#if defined(__gfx1250__) &&                                             \
    __has_builtin(__builtin_amdgcn_global_load_async_to_lds_b128) &&    \
    __has_builtin(__builtin_amdgcn_s_wait_asynccnt)
#define USE_ASYNC_LDS 1
#else
#define USE_ASYNC_LDS 0
#endif

#if USE_ASYNC_LDS
// Builtin signature (from toolchain diagnostic): the b128 payload is
// 'int __attribute__((vector_size(16)))', pointers in AS1 (global) / AS3 (LDS).
typedef int b128_t __attribute__((vector_size(16)));
typedef __attribute__((address_space(1))) b128_t* g_b128_p;
typedef __attribute__((address_space(3))) b128_t* l_b128_p;

__device__ __forceinline__ void async_cp16(const void* g, void* l) {
  __builtin_amdgcn_global_load_async_to_lds_b128(
      (g_b128_p)(unsigned long long)g,
      (l_b128_p)(unsigned long long)l,
      /*offset=*/0, /*cpol=*/0);
}
#endif

// ---------------------------------------------------------------------------
// WMMA helpers (CDNA5 gfx1250, wave32). Layouts per ISA 7.12.2.
// ---------------------------------------------------------------------------
__device__ __forceinline__ v8f wmma_bf16(v16bf a, v16bf b, v8f c) {
  // D = A(16x32 bf16) x B(32x16 bf16) + C(16x16 f32)
  return __builtin_amdgcn_wmma_f32_16x16x32_bf16(
      /*neg_a=*/false, a, /*neg_b=*/false, b,
      /*c_mod=*/(short)0, c, /*reuse_a=*/false, /*reuse_b=*/false);
}

// A-matrix fragment 16x32 bf16 from row-major [16][strideHalfs] buffer.
// lane<16: row=lane, k elements {0..7, 16..23}; lane>=16: row=lane-16, k+8.
__device__ __forceinline__ v16bf frag_a_ld(const __bf16* base, int strideHalfs, int lane) {
  const int r  = lane & 15;
  const int kh = (lane >> 4) << 3;               // 0 or 8 halfs
  const __bf16* p = base + r * strideHalfs + kh;
  v8bf lo = *(const v8bf*)(p);                   // k = kh .. kh+7
  v8bf hi = *(const v8bf*)(p + 16);              // k = kh+16 .. kh+23
  v16bf f;
#pragma unroll
  for (int e = 0; e < 8; ++e) { f[e] = lo[e]; f[e + 8] = hi[e]; }
  return f;
}

// B-matrix fragment 32x16 bf16 from an N-major buffer BT[n][k] ([16][strideHalfs]).
// lane<16: col=lane, k elements {0..7, 16..23}; lane>=16: col=lane-16, k+8.
__device__ __forceinline__ v16bf frag_b_ld(const __bf16* base, int strideHalfs, int lane) {
  const int n  = lane & 15;
  const int kh = (lane >> 4) << 3;               // 0 or 8 halfs
  const __bf16* p = base + n * strideHalfs + kh;
  v8bf lo = *(const v8bf*)(p);
  v8bf hi = *(const v8bf*)(p + 16);
  v16bf f;
#pragma unroll
  for (int e = 0; e < 8; ++e) { f[e] = lo[e]; f[e + 8] = hi[e]; }
  return f;
}

// ---------------------------------------------------------------------------
// fp32 -> bf16 convert (straight)
// ---------------------------------------------------------------------------
__global__ __launch_bounds__(256) void k_cvt(const float* __restrict__ in,
                                             __bf16* __restrict__ out, int n) {
  int i = blockIdx.x * 256 + threadIdx.x;
  if (i < n) out[i] = (__bf16)in[i];
}

// fp32 [K][N] -> bf16 [N][K] (transposed convert), 32x32 LDS tile
__global__ __launch_bounds__(256) void k_cvt_t(const float* __restrict__ W,
                                               __bf16* __restrict__ WT, int K, int N) {
  __shared__ float t[32][33];
  const int bn = blockIdx.x * 32, bk = blockIdx.y * 32;
  const int tx = threadIdx.x, ty = threadIdx.y;   // 32 x 8
#pragma unroll
  for (int i = 0; i < 32; i += 8)
    t[ty + i][tx] = W[(size_t)(bk + ty + i) * N + bn + tx];
  __syncthreads();
#pragma unroll
  for (int i = 0; i < 32; i += 8)
    WT[(size_t)(bn + ty + i) * K + bk + tx] = (__bf16)t[tx][ty + i];
}

// ---------------------------------------------------------------------------
// Generic bf16 WMMA GEMM:  C[M,N] = A[M,K] * BT[N,K]^T (+bias) (+epilogue)
// Block tile 128x64, 8 waves (4x2), wave tile 32x32, K-step 64.
// Double-buffered LDS; staging via GLOBAL_LOAD_ASYNC_TO_LDS_B128 when
// available (6 async 16B chunks in flight per thread per stage).
// mode 0: f32 out   mode 1: relu -> bf16 out   mode 2: f32 out + residual
// ---------------------------------------------------------------------------
__global__ __launch_bounds__(256) void k_gemm_bf16(
    const __bf16* __restrict__ A, const __bf16* __restrict__ BT,
    const float* __restrict__ bias, const float* __restrict__ resid,
    float* __restrict__ outF, __bf16* __restrict__ outH,
    int M, int N, int K, int mode)
{
  __shared__ __bf16 ldsA[2][128 * 72];   // stride 72 halfs = 144B (16B aligned)
  __shared__ __bf16 ldsB[2][64 * 72];
  const int tid  = threadIdx.x;
  const int lane = tid & 31;
  const int wave = tid >> 5;
  const int wm = wave & 3;                  // 0..3 -> 32-row subtile
  const int wn = wave >> 2;                 // 0..1 -> 32-col subtile
  const int m0 = blockIdx.y * 128;
  const int n0 = blockIdx.x * 64;

  v8f acc[2][2] = {};

  // Stage tile (k0) into LDS buffer p: A 128x64 (4 chunks/thread),
  // BT 64x64 (2 chunks/thread); 16B per chunk per lane.
  auto stage = [&](int p, int k0) {
#pragma unroll
    for (int i = 0; i < 4; ++i) {
      const int c = tid + i * 256;
      const int row = c >> 3, col = (c & 7) * 8;
#if USE_ASYNC_LDS
      async_cp16(&A[(size_t)(m0 + row) * K + k0 + col], &ldsA[p][row * 72 + col]);
#else
      *(v8bf*)&ldsA[p][row * 72 + col] =
          *(const v8bf*)&A[(size_t)(m0 + row) * K + k0 + col];
#endif
    }
#pragma unroll
    for (int i = 0; i < 2; ++i) {
      const int c = tid + i * 256;
      const int row = c >> 3, col = (c & 7) * 8;
#if USE_ASYNC_LDS
      async_cp16(&BT[(size_t)(n0 + row) * K + k0 + col], &ldsB[p][row * 72 + col]);
#else
      *(v8bf*)&ldsB[p][row * 72 + col] =
          *(const v8bf*)&BT[(size_t)(n0 + row) * K + k0 + col];
#endif
    }
  };

  const int T = K >> 6;                       // number of 64-wide K tiles
  stage(0, 0);                                // prologue

  for (int t = 0; t < T; ++t) {
    const int p = t & 1;
    if (t + 1 < T) {
      stage(p ^ 1, (t + 1) << 6);             // overlap DMA with WMMA on cur
#if USE_ASYNC_LDS
      // 12 may be in flight; async loads retire in order, so <=6 means the
      // current buffer's 6 chunks (per thread) have landed.
      __builtin_amdgcn_s_wait_asynccnt(6);
#endif
    } else {
#if USE_ASYNC_LDS
      __builtin_amdgcn_s_wait_asynccnt(0);
#endif
    }
    __syncthreads();                          // all waves' chunks visible
#pragma unroll
    for (int kc = 0; kc < 2; ++kc) {
      v16bf a0 = frag_a_ld(&ldsA[p][(wm * 32 +  0) * 72 + kc * 32], 72, lane);
      v16bf a1 = frag_a_ld(&ldsA[p][(wm * 32 + 16) * 72 + kc * 32], 72, lane);
      v16bf b0 = frag_b_ld(&ldsB[p][(wn * 32 +  0) * 72 + kc * 32], 72, lane);
      v16bf b1 = frag_b_ld(&ldsB[p][(wn * 32 + 16) * 72 + kc * 32], 72, lane);
      acc[0][0] = wmma_bf16(a0, b0, acc[0][0]);
      acc[0][1] = wmma_bf16(a0, b1, acc[0][1]);
      acc[1][0] = wmma_bf16(a1, b0, acc[1][0]);
      acc[1][1] = wmma_bf16(a1, b1, acc[1][1]);
    }
    __syncthreads();                          // buffer p free for overwrite
  }

  const int hi = lane >> 4;
#pragma unroll
  for (int i = 0; i < 2; ++i) {
#pragma unroll
    for (int j = 0; j < 2; ++j) {
      const int n = n0 + wn * 32 + j * 16 + (lane & 15);
      const float bv = bias ? bias[n] : 0.f;
#pragma unroll
      for (int r = 0; r < 8; ++r) {
        const int m = m0 + wm * 32 + i * 16 + r + hi * 8;
        const size_t idx = (size_t)m * N + n;
        const float v = acc[i][j][r] + bv;
        if (mode == 0)      outF[idx] = v;
        else if (mode == 1) outH[idx] = (__bf16)fmaxf(v, 0.f);
        else                outF[idx] = v + resid[idx];
      }
    }
  }
}

// ---------------------------------------------------------------------------
// Q/K/V reshape: f32 [S*B][D] -> bf16 per-head layouts.
// Q,K: [head][s][dh] (A/B row-major), V: [head][dh][s] (d-major for P*V B-frag)
// Q scaled by 1/sqrt(dh).
// ---------------------------------------------------------------------------
__global__ __launch_bounds__(256) void k_qkv_reshape(
    const float* __restrict__ Qf, const float* __restrict__ Kf,
    const float* __restrict__ Vf,
    __bf16* __restrict__ Qb, __bf16* __restrict__ Kb, __bf16* __restrict__ VbT)
{
  const int idx = blockIdx.x * 256 + threadIdx.x;      // over 4096*1024
  const int row = idx >> 10, col = idx & 1023;
  const int s = row >> 1, b = row & 1;                 // B = 2
  const int h = col >> 6, d = col & 63;                // dh = 64
  const int n = b * 16 + h;                            // head id, H = 16
  Qb [((size_t)n * 2048 + s) * 64 + d] = (__bf16)(Qf[idx] * 0.125f); // 1/sqrt(64)
  Kb [((size_t)n * 2048 + s) * 64 + d] = (__bf16)Kf[idx];
  VbT[((size_t)n * 64 + d) * 2048 + s] = (__bf16)Vf[idx];
}

// ---------------------------------------------------------------------------
// Flash-style attention. One wave = one (head, 16-query tile).
// Per 32-key chunk: 4 WMMAs for scores, online softmax (16-lane shfl
// reductions), P through LDS (C-layout -> A-layout), 4 WMMAs for P*V.
// ---------------------------------------------------------------------------
__global__ __launch_bounds__(256) void k_attention(
    const __bf16* __restrict__ Qb, const __bf16* __restrict__ Kb,
    const __bf16* __restrict__ VbT, __bf16* __restrict__ Ob)
{
  __shared__ __bf16 P[8][16 * 40];                   // per-wave P tile, stride 40
  const int tid = threadIdx.x, lane = tid & 31, wave = tid >> 5;
  const int w    = blockIdx.x * 8 + wave;            // 4096 waves total
  const int head = w >> 7;                           // 32 heads x 128 tiles
  const int s0   = (w & 127) * 16;
  const int b    = head >> 4, h = head & 15;
  const int hi   = lane >> 4;
  const int S    = 2048;
  const __bf16* Qh = Qb  + (size_t)head * S * 64;
  const __bf16* Kh = Kb  + (size_t)head * S * 64;
  const __bf16* Vh = VbT + (size_t)head * 64 * S;
  __bf16* Pw = &P[wave][0];

  const v16bf qa0 = frag_a_ld(Qh + s0 * 64 +  0, 64, lane);  // d 0..31
  const v16bf qa1 = frag_a_ld(Qh + s0 * 64 + 32, 64, lane);  // d 32..63

  v8f oacc[4] = {};
  float mrun[8], lrun[8];
#pragma unroll
  for (int r = 0; r < 8; ++r) { mrun[r] = -1e30f; lrun[r] = 0.f; }

  for (int t0 = 0; t0 < S; t0 += 32) {
    v8f sc[2] = {};
#pragma unroll
    for (int j = 0; j < 2; ++j) {                    // two 16-key subtiles
      const __bf16* Kt = Kh + (size_t)(t0 + j * 16) * 64;
      v16bf kb0 = frag_b_ld(Kt +  0, 64, lane);      // d 0..31
      v16bf kb1 = frag_b_ld(Kt + 32, 64, lane);      // d 32..63
      sc[j] = wmma_bf16(qa0, kb0, sc[j]);
      sc[j] = wmma_bf16(qa1, kb1, sc[j]);
    }
    // online softmax: rows r+8*hi live across the 16 lanes of each half-wave
    float p0[8], p1[8], scl[8];
#pragma unroll
    for (int r = 0; r < 8; ++r) {
      float mx = fmaxf(sc[0][r], sc[1][r]);
#pragma unroll
      for (int off = 8; off >= 1; off >>= 1)
        mx = fmaxf(mx, __shfl_xor(mx, off, 32));
      const float mnew = fmaxf(mrun[r], mx);
      scl[r] = __expf(mrun[r] - mnew);
      p0[r]  = __expf(sc[0][r] - mnew);
      p1[r]  = __expf(sc[1][r] - mnew);
      float rs = p0[r] + p1[r];
#pragma unroll
      for (int off = 8; off >= 1; off >>= 1)
        rs += __shfl_xor(rs, off, 32);
      lrun[r] = lrun[r] * scl[r] + rs;
      mrun[r] = mnew;
    }
#pragma unroll
    for (int dt = 0; dt < 4; ++dt)
#pragma unroll
      for (int r = 0; r < 8; ++r) oacc[dt][r] *= scl[r];

    // P (C-layout) -> LDS row-major [16][32] -> A-fragment
    const int tl = lane & 15;
#pragma unroll
    for (int r = 0; r < 8; ++r) {
      Pw[(r + hi * 8) * 40 + tl]      = (__bf16)p0[r];
      Pw[(r + hi * 8) * 40 + 16 + tl] = (__bf16)p1[r];
    }
    const v16bf pa = frag_a_ld(Pw, 40, lane);        // 16 x 32 (k = t-local)
#pragma unroll
    for (int dt = 0; dt < 4; ++dt) {
      v16bf vb = frag_b_ld(Vh + (size_t)(dt * 16) * S + t0, S, lane);
      oacc[dt] = wmma_bf16(pa, vb, oacc[dt]);
    }
  }

  float inv[8];
#pragma unroll
  for (int r = 0; r < 8; ++r) inv[r] = 1.f / lrun[r];
#pragma unroll
  for (int dt = 0; dt < 4; ++dt) {
    const int col = h * 64 + dt * 16 + (lane & 15);
#pragma unroll
    for (int r = 0; r < 8; ++r) {
      const int s = s0 + r + hi * 8;
      Ob[((size_t)s * 2 + b) * 1024 + col] = (__bf16)(oacc[dt][r] * inv[r]);
    }
  }
}

// ---------------------------------------------------------------------------
// LayerNorm over D=1024, one block per row. Writes f32 and optional bf16.
// ---------------------------------------------------------------------------
__global__ __launch_bounds__(256) void k_layernorm(
    const float* __restrict__ in, const float* __restrict__ gamma,
    const float* __restrict__ beta, float* __restrict__ outF,
    __bf16* __restrict__ outH, int D)
{
  __shared__ float rs[256], rs2[256];
  const int row = blockIdx.x, tid = threadIdx.x;
  const float* x = in + (size_t)row * D;
  float s = 0.f, s2 = 0.f;
  for (int i = tid; i < D; i += 256) { float v = x[i]; s += v; s2 += v * v; }
  rs[tid] = s; rs2[tid] = s2;
  __syncthreads();
  for (int st = 128; st > 0; st >>= 1) {
    if (tid < st) { rs[tid] += rs[tid + st]; rs2[tid] += rs2[tid + st]; }
    __syncthreads();
  }
  const float mean = rs[0] / D;
  const float var  = rs2[0] / D - mean * mean;
  const float rstd = rsqrtf(var + 1e-5f);
  for (int i = tid; i < D; i += 256) {
    const float v = (x[i] - mean) * rstd * gamma[i] + beta[i];
    outF[(size_t)row * D + i] = v;
    if (outH) outH[(size_t)row * D + i] = (__bf16)v;
  }
}

// ---------------------------------------------------------------------------
// Host orchestration
// ---------------------------------------------------------------------------
extern "C" void kernel_launch(void* const* d_in, const int* in_sizes, int n_in,
                              void* d_out, int out_size, void* d_ws, size_t ws_size,
                              hipStream_t stream) {
  const float* x   = (const float*)d_in[0];
  const float* Wq  = (const float*)d_in[1];
  const float* bq  = (const float*)d_in[2];
  const float* Wk  = (const float*)d_in[3];
  const float* bk  = (const float*)d_in[4];
  const float* Wv  = (const float*)d_in[5];
  const float* bv  = (const float*)d_in[6];
  const float* Wo  = (const float*)d_in[7];
  const float* bo  = (const float*)d_in[8];
  const float* g1  = (const float*)d_in[9];
  const float* b1  = (const float*)d_in[10];
  const float* W1  = (const float*)d_in[11];
  const float* bb1 = (const float*)d_in[12];
  const float* W2  = (const float*)d_in[13];
  const float* bb2 = (const float*)d_in[14];
  const float* g2  = (const float*)d_in[15];
  const float* b2  = (const float*)d_in[16];

  constexpr int S = 2048, B = 2, D = 1024, F = 4096;
  constexpr int M = S * B;                      // 4096 rows

  char* ws = (char*)d_ws;
  size_t off = 0;
  auto arena = [&](size_t bytes) {
    size_t r = off; off = (off + bytes + 255) & ~(size_t)255; return r;
  };
  __bf16* xb   = (__bf16*)(ws + arena((size_t)M * D * 2));
  __bf16* WqT  = (__bf16*)(ws + arena((size_t)D * D * 2));
  __bf16* WkT  = (__bf16*)(ws + arena((size_t)D * D * 2));
  __bf16* WvT  = (__bf16*)(ws + arena((size_t)D * D * 2));
  __bf16* WoT  = (__bf16*)(ws + arena((size_t)D * D * 2));
  __bf16* W1T  = (__bf16*)(ws + arena((size_t)F * D * 2));   // [F][D]
  __bf16* W2T  = (__bf16*)(ws + arena((size_t)D * F * 2));   // [D][F]
  __bf16* Qb   = (__bf16*)(ws + arena((size_t)M * D * 2));
  __bf16* Kb   = (__bf16*)(ws + arena((size_t)M * D * 2));
  __bf16* VbT  = (__bf16*)(ws + arena((size_t)M * D * 2));
  __bf16* Ob   = (__bf16*)(ws + arena((size_t)M * D * 2));
  float*  Qf   = (float*)(ws + arena((size_t)M * D * 4));
  float*  Kf   = (float*)(ws + arena((size_t)M * D * 4));
  float*  Vf   = (float*)(ws + arena((size_t)M * D * 4));
  __bf16* x1b  = (__bf16*)(ws + arena((size_t)M * D * 2));
  __bf16* hb   = (__bf16*)(ws + arena((size_t)M * F * 2));
  // f32 scratch reuse after QKV reshape frees Qf/Kf/Vf:
  float* y1 = Qf;   // x + attn_out @ Wo
  float* x1 = Kf;   // LN1 output (f32, kept for final residual)
  float* y2 = Vf;   // x1 + ffn

  const dim3 blk(256);

  // 1) converts / transposed weight converts
  k_cvt<<<(M * D) / 256, blk, 0, stream>>>(x, xb, M * D);
  {
    dim3 tb(32, 8);
    k_cvt_t<<<dim3(D / 32, D / 32), tb, 0, stream>>>(Wq, WqT, D, D);
    k_cvt_t<<<dim3(D / 32, D / 32), tb, 0, stream>>>(Wk, WkT, D, D);
    k_cvt_t<<<dim3(D / 32, D / 32), tb, 0, stream>>>(Wv, WvT, D, D);
    k_cvt_t<<<dim3(D / 32, D / 32), tb, 0, stream>>>(Wo, WoT, D, D);
    k_cvt_t<<<dim3(F / 32, D / 32), tb, 0, stream>>>(W1, W1T, D, F); // -> [F][D]
    k_cvt_t<<<dim3(D / 32, F / 32), tb, 0, stream>>>(W2, W2T, F, D); // -> [D][F]
  }

  // 2) Q/K/V projections (f32 out)
  k_gemm_bf16<<<dim3(D / 64, M / 128), blk, 0, stream>>>(
      xb, WqT, bq, nullptr, Qf, nullptr, M, D, D, 0);
  k_gemm_bf16<<<dim3(D / 64, M / 128), blk, 0, stream>>>(
      xb, WkT, bk, nullptr, Kf, nullptr, M, D, D, 0);
  k_gemm_bf16<<<dim3(D / 64, M / 128), blk, 0, stream>>>(
      xb, WvT, bv, nullptr, Vf, nullptr, M, D, D, 0);

  // 3) reshape to head layouts (Q scaled by 1/sqrt(dh))
  k_qkv_reshape<<<(M * D) / 256, blk, 0, stream>>>(Qf, Kf, Vf, Qb, Kb, VbT);

  // 4) attention (4096 waves, 8 per block)
  k_attention<<<512, blk, 0, stream>>>(Qb, Kb, VbT, Ob);

  // 5) output projection + residual: y1 = x + Ob@Wo + bo
  k_gemm_bf16<<<dim3(D / 64, M / 128), blk, 0, stream>>>(
      Ob, WoT, bo, x, y1, nullptr, M, D, D, 2);

  // 6) LN1 -> x1 (f32) and x1b (bf16)
  k_layernorm<<<M, blk, 0, stream>>>(y1, g1, b1, x1, x1b, D);

  // 7) FFN1: h = relu(x1 @ W1 + bb1) -> bf16
  k_gemm_bf16<<<dim3(F / 64, M / 128), blk, 0, stream>>>(
      x1b, W1T, bb1, nullptr, nullptr, hb, M, F, D, 1);

  // 8) FFN2 + residual: y2 = x1 + h @ W2 + bb2
  k_gemm_bf16<<<dim3(D / 64, M / 128), blk, 0, stream>>>(
      hb, W2T, bb2, x1, y2, nullptr, M, D, F, 2);

  // 9) LN2 -> d_out (f32)
  k_layernorm<<<M, blk, 0, stream>>>(y2, g2, b2, (float*)d_out, nullptr, D);
}
